// TransitionDown_43782896615728
// MI455X (gfx1250) — matrix operand order repaired and checked by reading, not compiled, not run on hardware
//
#include <hip/hip_runtime.h>
#include <hip/hip_bf16.h>
#include <stdint.h>

// Problem constants (match reference setup_inputs)
#define B_   8
#define N_   4096
#define D_   64
#define K_   1024
#define NN_  16
#define C_   128
#define M_   (B_*K_*NN_)        // 131072 rows through the MLP
#define LDA1 96                 // K=67 padded to 3*32 for bf16 WMMA
#define KS1  3                  // 96/32 k-steps, layer 1
#define KS2  4                  // 128/32 k-steps, layer 2
#define NT_  8                  // 128/16 n-tiles
#define TILES_M (M_/16)         // 8192 = one block per m-tile
#define EPSV 1e-5f

typedef __attribute__((ext_vector_type(16))) __bf16 v16bf;
typedef __attribute__((ext_vector_type(8)))  float  v8f;

static __device__ __forceinline__ unsigned short f2bf(float f) {
  union { float f; unsigned u; } x; x.f = f;
  unsigned u = x.u;
  unsigned r = (u + 0x7FFFu + ((u >> 16) & 1u)) >> 16;   // RNE
  return (unsigned short)r;
}
static __device__ __forceinline__ float bf2f(unsigned short h) {
  union { unsigned u; float f; } x; x.u = ((unsigned)h) << 16;
  return x.f;
}

// ---------------------------------------------------------------------------
// 1) Farthest point sampling: one 1024-thread block per batch, dist[] in LDS.
// ---------------------------------------------------------------------------
__global__ __launch_bounds__(1024)
void fps_kernel(const float* __restrict__ xyz, int* __restrict__ fps_idx) {
  __shared__ float dist[N_];
  __shared__ float rv[1024];
  __shared__ int   ri[1024];
  __shared__ float cen[3];
  __shared__ int   sFar;
  const int b = blockIdx.x;
  const int t = threadIdx.x;
  const float* xb = xyz + (size_t)b * N_ * 3;
  for (int n = t; n < N_; n += 1024) dist[n] = 1e10f;
  if (t == 0) sFar = 0;
  __syncthreads();
  for (int i = 0; i < K_; ++i) {
    if (t == 0) {
      int far = sFar;
      fps_idx[b * K_ + i] = far;              // scan emits previous 'far'
      cen[0] = xb[far*3+0]; cen[1] = xb[far*3+1]; cen[2] = xb[far*3+2];
    }
    __syncthreads();
    const float cx = cen[0], cy = cen[1], cz = cen[2];
    float bestv = -1.0f; int besti = 0;
    for (int n = t; n < N_; n += 1024) {
      float dx = xb[n*3+0]-cx, dy = xb[n*3+1]-cy, dz = xb[n*3+2]-cz;
      float d  = dx*dx + dy*dy + dz*dz;
      float o  = dist[n];
      d = d < o ? d : o;
      dist[n] = d;
      if (d > bestv) { bestv = d; besti = n; }   // strict > => lowest index
    }
    rv[t] = bestv; ri[t] = besti;
    __syncthreads();
    for (int s = 512; s > 0; s >>= 1) {
      if (t < s) {
        float v2 = rv[t+s]; int i2 = ri[t+s];
        if (v2 > rv[t] || (v2 == rv[t] && i2 < ri[t])) { rv[t] = v2; ri[t] = i2; }
      }
      __syncthreads();
    }
    if (t == 0) sFar = ri[0];
    __syncthreads();
  }
}

// ---------------------------------------------------------------------------
// 2) new_xyz = gather(xyz, fps_idx) -> first 24576 floats of d_out
// ---------------------------------------------------------------------------
__global__ void newxyz_kernel(const float* __restrict__ xyz,
                              const int* __restrict__ fps_idx,
                              float* __restrict__ out) {
  int i = blockIdx.x * blockDim.x + threadIdx.x;
  if (i >= B_ * K_ * 3) return;
  int c = i % 3, bk = i / 3, b = bk / K_, kk = bk % K_;
  int p = fps_idx[b * K_ + kk];
  out[i] = xyz[((size_t)b * N_ + p) * 3 + c];
}

// ---------------------------------------------------------------------------
// 3) kNN: one 256-thread block per (b,k); distances in LDS; 16 exact argmins
//    using the same expanded |c|^2+|x|^2-2c.x formula as the reference.
// ---------------------------------------------------------------------------
__global__ __launch_bounds__(256)
void knn_kernel(const float* __restrict__ xyz, const int* __restrict__ fps_idx,
                int* __restrict__ knn_idx) {
  __shared__ float d[N_];
  __shared__ float rv[256];
  __shared__ int   ri[256];
  const int bk = blockIdx.x, b = bk / K_, t = threadIdx.x;
  const float* xb = xyz + (size_t)b * N_ * 3;
  const int p = fps_idx[bk];
  const float cx = xb[p*3+0], cy = xb[p*3+1], cz = xb[p*3+2];
  const float cc = cx*cx + cy*cy + cz*cz;
  for (int n = t; n < N_; n += 256) {
    float x = xb[n*3+0], y = xb[n*3+1], z = xb[n*3+2];
    d[n] = cc + x*x + y*y + z*z - 2.0f*(cx*x + cy*y + cz*z);
  }
  __syncthreads();
  for (int j = 0; j < NN_; ++j) {
    float bv = __builtin_inff(); int bi = 0;
    for (int n = t; n < N_; n += 256) {
      float v = d[n];
      if (v < bv) { bv = v; bi = n; }
    }
    rv[t] = bv; ri[t] = bi;
    __syncthreads();
    for (int s = 128; s > 0; s >>= 1) {
      if (t < s) {
        float v2 = rv[t+s]; int i2 = ri[t+s];
        if (v2 < rv[t] || (v2 == rv[t] && i2 < ri[t])) { rv[t] = v2; ri[t] = i2; }
      }
      __syncthreads();
    }
    if (t == 0) { knn_idx[(size_t)bk * NN_ + j] = ri[0]; d[ri[0]] = 3.0e38f; }
    __syncthreads();
  }
}

// ---------------------------------------------------------------------------
// 4) Build h = [xyz_rel(3) | feats(64) | zero-pad to 96] as bf16, row-major.
// ---------------------------------------------------------------------------
__global__ void buildh_kernel(const float* __restrict__ xyz,
                              const float* __restrict__ points,
                              const int* __restrict__ fps_idx,
                              const int* __restrict__ knn_idx,
                              unsigned short* __restrict__ hbuf) {
  const int r = blockIdx.x;          // row in [0, M_)
  const int c = threadIdx.x;         // 0..95
  const int b  = r / (K_ * NN_);
  const int kk = (r % (K_ * NN_)) / NN_;
  const int p  = knn_idx[r];
  float v = 0.f;
  if (c < 3) {
    int ctr = fps_idx[b * K_ + kk];
    v = xyz[((size_t)b * N_ + p) * 3 + c] - xyz[((size_t)b * N_ + ctr) * 3 + c];
  } else if (c < 67) {
    v = points[((size_t)b * N_ + p) * D_ + (c - 3)];
  }
  hbuf[(size_t)r * LDA1 + c] = f2bf(v);
}

// ---------------------------------------------------------------------------
// 5) Pack weights W[N=128][Kdim] into the exact bf16 B-matrix lane layout:
//    packed[((nt*ksteps+ks)*32+lane)*16+e] = W[nt*16+lane%16][ks*32+(lane/16)*8+e+(e>=8?8:0)]
// ---------------------------------------------------------------------------
__global__ void packw_kernel(const float* __restrict__ W,
                             unsigned short* __restrict__ out,
                             int Kdim, int ksteps) {
  int idx = blockIdx.x * blockDim.x + threadIdx.x;
  int total = NT_ * ksteps * 32 * 16;
  if (idx >= total) return;
  int e    = idx & 15;
  int lane = (idx >> 4) & 31;
  int ks   = (idx >> 9) % ksteps;
  int nt   = idx / (ksteps * 512);
  int n = nt * 16 + (lane & 15);
  int k = ks * 32 + (lane >> 4) * 8 + e + (e >= 8 ? 8 : 0);
  float v = (k < Kdim) ? W[(size_t)n * Kdim + k] : 0.f;
  out[idx] = f2bf(v);
}

// ---------------------------------------------------------------------------
// 6) Pack BN1 scale/shift into the A-operand lane layout used by GEMM2, so
//    each lane reads one contiguous 64B float16-vector per k-step.
//    Element e of lane L at k-step ks maps to channel
//      k = ks*32 + (L>>4)*8 + e + (e>=8 ? 8 : 0)
// ---------------------------------------------------------------------------
__global__ void packss_kernel(const float* __restrict__ ss,
                              float* __restrict__ scp, float* __restrict__ shp) {
  int idx = blockIdx.x * blockDim.x + threadIdx.x;   // KS2*32*16 = 2048
  if (idx >= KS2 * 32 * 16) return;
  int e    = idx & 15;
  int lane = (idx >> 4) & 31;
  int ks   = idx >> 9;
  int k = ks * 32 + (lane >> 4) * 8 + e + (e >= 8 ? 8 : 0);
  scp[idx] = ss[k];
  shp[idx] = ss[C_ + k];
}

// ---------------------------------------------------------------------------
// 7) WMMA GEMM with fused epilogue. One block per 16-row m-tile; its 8 waves
//    cover all 8 n-tiles (full 128 channels), so the block computes exact
//    per-channel partial sums/sumsq of (acc+bias) for BatchNorm: lane pairs
//    (l, l^16) hold the two row-halves of a column and combine via shfl_xor.
//    APPLY:    transform A on the fly: relu(a*scale+shift) (fused BN1+ReLU)
//    OUT_BF16: store result rounded to bf16 (feeds next bf16 GEMM)
// ---------------------------------------------------------------------------
template <bool APPLY, bool OUT_BF16>
__global__ __launch_bounds__(256)
void gemm_wmma_kernel(const unsigned short* __restrict__ A,
                      const unsigned short* __restrict__ Bp,
                      const float* __restrict__ bias,
                      const float* __restrict__ scp,
                      const float* __restrict__ shp,
                      float* __restrict__ Cf32,
                      unsigned short* __restrict__ Cb16,
                      float* __restrict__ ps, float* __restrict__ pq,
                      int lda, int ksteps) {
  const int tm   = blockIdx.x;
  const int tn   = threadIdx.x >> 5;   // wave id = n-tile
  const int lane = threadIdx.x & 31;
  const int half = lane >> 4;          // 0: K 0-7/16-23, 1: K 8-15/24-31
  const int l16  = lane & 15;
  v8f acc = {};
  const unsigned short* arow  = A  + (size_t)(tm * 16 + l16) * lda + half * 8;
  const unsigned short* bbase = Bp + ((size_t)tn * ksteps * 32 + lane) * 16;
  for (int ks = 0; ks < ksteps; ++ks) {
    union { uint4 q[2]; unsigned short s[16]; v16bf v; } ua;
    union { uint4 q[2]; v16bf v; } ub;
    const uint4* ap = (const uint4*)(arow + ks * 32);
    ua.q[0] = ap[0];               // K = kbase .. kbase+7
    ua.q[1] = ap[2];               // K = kbase+16 .. kbase+23 (+32 bytes)
    if (APPLY) {                   // fused BN1 + ReLU on the A operand
      const float* sc = scp + (size_t)(ks * 32 + lane) * 16;
      const float* sh = shp + (size_t)(ks * 32 + lane) * 16;
#pragma unroll
      for (int e = 0; e < 16; ++e) {
        float v = fmaf(bf2f(ua.s[e]), sc[e], sh[e]);
        v = v > 0.f ? v : 0.f;
        ua.s[e] = f2bf(v);
      }
    }
    const uint4* bp = (const uint4*)(bbase + (size_t)ks * 512);
    ub.q[0] = bp[0];
    ub.q[1] = bp[1];
    acc = __builtin_amdgcn_wmma_f32_16x16x32_bf16(
        false, ua.v, false, ub.v, (short)0, acc, false, false);
  }
  const float bv = bias[tn * 16 + l16];
  const size_t base = ((size_t)tm * 16 + half * 8) * C_ + tn * 16 + l16;
  float s = 0.f, q = 0.f;
#pragma unroll
  for (int r = 0; r < 8; ++r) {
    float v = acc[r] + bv;
    s += v; q += v * v;
    if (OUT_BF16) Cb16[base + (size_t)r * C_] = f2bf(v);
    else          Cf32[base + (size_t)r * C_] = v;
  }
  // combine the two 8-row halves of each column (lane <-> lane^16)
  s += __shfl_xor(s, 16, 32);
  q += __shfl_xor(q, 16, 32);
  if (lane < 16) {
    ps[(size_t)tm * C_ + tn * 16 + l16] = s;
    pq[(size_t)tm * C_ + tn * 16 + l16] = q;
  }
}

// ---------------------------------------------------------------------------
// 8) BN finalize: fixed-order reduction of the 8192 block partials
//    (deterministic across graph replays), emits per-channel scale/shift.
// ---------------------------------------------------------------------------
__global__ void bn_finalize_kernel(const float* __restrict__ ps,
                                   const float* __restrict__ pq,
                                   const float* __restrict__ g,
                                   const float* __restrict__ beta,
                                   float* __restrict__ ss) {
  const int c = threadIdx.x;
  float s = 0.f, q = 0.f;
  for (int i = 0; i < TILES_M; ++i) { s += ps[(size_t)i * C_ + c]; q += pq[(size_t)i * C_ + c]; }
  float mean = s / (float)M_;
  float var  = q / (float)M_ - mean * mean;
  float sc = g[c] * rsqrtf(var + EPSV);
  ss[c]      = sc;
  ss[C_ + c] = beta[c] - mean * sc;
}

// ---------------------------------------------------------------------------
// 9) Fused BN2+ReLU+max over the NN=16 neighbors -> new_points in d_out.
// ---------------------------------------------------------------------------
__global__ void maxpool_kernel(const float* __restrict__ y2,
                               const float* __restrict__ ss,
                               float* __restrict__ out) {
  const int bk = blockIdx.x;         // (b*K + k)
  const int c  = threadIdx.x;        // channel
  const size_t r0 = (size_t)bk * NN_;
  const float sc = ss[c], sh = ss[C_ + c];
  float mx = -3.4e38f;
  for (int j = 0; j < NN_; ++j) {
    float v = y2[(r0 + j) * C_ + c] * sc + sh;
    v = v > 0.f ? v : 0.f;
    if (v > mx) mx = v;
  }
  out[(size_t)B_ * K_ * 3 + (size_t)bk * C_ + c] = mx;
}

// ---------------------------------------------------------------------------
extern "C" void kernel_launch(void* const* d_in, const int* in_sizes, int n_in,
                              void* d_out, int out_size, void* d_ws, size_t ws_size,
                              hipStream_t stream) {
  (void)in_sizes; (void)n_in; (void)out_size; (void)ws_size;
  const float* xyz    = (const float*)d_in[0];
  const float* points = (const float*)d_in[1];
  const float* W1     = (const float*)d_in[2];
  const float* b1     = (const float*)d_in[3];
  const float* g1     = (const float*)d_in[4];
  const float* be1    = (const float*)d_in[5];
  const float* W2     = (const float*)d_in[6];
  const float* b2     = (const float*)d_in[7];
  const float* g2     = (const float*)d_in[8];
  const float* be2    = (const float*)d_in[9];
  float* out = (float*)d_out;

  char* w = (char*)d_ws;
  auto alloc = [&](size_t bytes) -> char* {
    char* p = w; w += (bytes + 255) & ~(size_t)255; return p;
  };
  int*            fps_idx = (int*)alloc((size_t)B_ * K_ * 4);
  int*            knn_idx = (int*)alloc((size_t)B_ * K_ * NN_ * 4);
  unsigned short* hbuf    = (unsigned short*)alloc((size_t)M_ * LDA1 * 2);
  unsigned short* w1p     = (unsigned short*)alloc((size_t)NT_ * KS1 * 512 * 2);
  unsigned short* w2p     = (unsigned short*)alloc((size_t)NT_ * KS2 * 512 * 2);
  unsigned short* y1      = (unsigned short*)alloc((size_t)M_ * C_ * 2);  // bf16
  float*          y2      = (float*)alloc((size_t)M_ * C_ * 4);
  float*          ps      = (float*)alloc((size_t)TILES_M * C_ * 4);
  float*          pq      = (float*)alloc((size_t)TILES_M * C_ * 4);
  float*          ss1     = (float*)alloc(2 * C_ * 4);
  float*          ss2     = (float*)alloc(2 * C_ * 4);
  float*          scp     = (float*)alloc((size_t)KS2 * 32 * 16 * 4);
  float*          shp     = (float*)alloc((size_t)KS2 * 32 * 16 * 4);

  fps_kernel<<<B_, 1024, 0, stream>>>(xyz, fps_idx);
  newxyz_kernel<<<(B_ * K_ * 3 + 255) / 256, 256, 0, stream>>>(xyz, fps_idx, out);
  knn_kernel<<<B_ * K_, 256, 0, stream>>>(xyz, fps_idx, knn_idx);
  buildh_kernel<<<M_, 96, 0, stream>>>(xyz, points, fps_idx, knn_idx, hbuf);
  packw_kernel<<<(NT_ * KS1 * 512 + 255) / 256, 256, 0, stream>>>(W1, w1p, 67, KS1);
  packw_kernel<<<(NT_ * KS2 * 512 + 255) / 256, 256, 0, stream>>>(W2, w2p, 128, KS2);

  // Layer 1: GEMM + bias, bf16 output, fused BN1 partial stats
  gemm_wmma_kernel<false, true><<<TILES_M, 256, 0, stream>>>(
      hbuf, w1p, b1, nullptr, nullptr, nullptr, y1, ps, pq, LDA1, KS1);
  bn_finalize_kernel<<<1, C_, 0, stream>>>(ps, pq, g1, be1, ss1);
  packss_kernel<<<(KS2 * 32 * 16 + 255) / 256, 256, 0, stream>>>(ss1, scp, shp);

  // Layer 2: fused BN1+ReLU on A, GEMM + bias, f32 output, fused BN2 stats
  gemm_wmma_kernel<true, false><<<TILES_M, 256, 0, stream>>>(
      y1, w2p, b2, scp, shp, y2, nullptr, ps, pq, C_, KS2);
  bn_finalize_kernel<<<1, C_, 0, stream>>>(ps, pq, g2, be2, ss2);

  maxpool_kernel<<<B_ * K_, C_, 0, stream>>>(y2, ss2, out);
}
